// LlamaAttention_77163382440158
// MI455X (gfx1250) — compile-verified, matrix operand order checked by reading, not run on hardware
//
#include <hip/hip_runtime.h>
#include <hip/hip_bf16.h>
#include <math.h>

// ---------------------------------------------------------------------------
// MI455X (gfx1250) Llama attention block, all matmuls via v_wmma_f32_16x16x32_bf16.
// Weights pre-transposed/converted to bf16 [N][K] so the GEMM inner loop is
// pure b128 copies + WMMA (no conversions, no LDS scatter).
// ---------------------------------------------------------------------------

typedef __attribute__((ext_vector_type(16))) __bf16 v16bf;
typedef __attribute__((ext_vector_type(8)))  __bf16 v8bf;
typedef __attribute__((ext_vector_type(4)))  __bf16 v4bf;
typedef __attribute__((ext_vector_type(8)))  float  v8f;
typedef __attribute__((ext_vector_type(4)))  float  v4f;

#define SEQ          2048
#define HIDDEN       4096
#define NUM_HEADS    32
#define NUM_KV_HEADS 8
#define HEAD_DIM     128
#define GROUP        (NUM_HEADS / NUM_KV_HEADS)          // 4
#define Q_SIZE       (NUM_HEADS * HEAD_DIM)              // 4096
#define KV_SIZE      (NUM_KV_HEADS * HEAD_DIM)           // 1024
#define QKV_OUT      (Q_SIZE + 2 * KV_SIZE)              // 6144

// ISA 7.12.2 (16-bit A/B frag): lane holds m/n = lane&15, kbase = (lane>>4)*8,
// elements = contiguous runs [kbase, kbase+8) and [kbase+16, kbase+24).
__device__ __forceinline__ v16bf frag16(const __bf16* row, int kb) {
  v8bf lo = *(const v8bf*)(row + kb);
  v8bf hi = *(const v8bf*)(row + kb + 16);
  return __builtin_shufflevector(lo, hi, 0, 1, 2, 3, 4, 5, 6, 7,
                                 8, 9, 10, 11, 12, 13, 14, 15);
}

__device__ __forceinline__ v8f wmma_bf16(v16bf a, v16bf b, v8f c) {
  return __builtin_amdgcn_wmma_f32_16x16x32_bf16(
      /*neg_a=*/false, a, /*neg_b=*/false, b,
      /*c_mod=*/(short)0, c, /*reuse_a=*/false, /*reuse_b=*/false);
}

// ---------------------------------------------------------------------------
// Elementwise f32 -> bf16 (8 elems/thread, b128 in / b128 out)
// ---------------------------------------------------------------------------
__global__ __launch_bounds__(256)
void convert_f32_bf16(const float* __restrict__ in, __bf16* __restrict__ out) {
  size_t i = ((size_t)blockIdx.x * 256 + threadIdx.x) * 8;
  v4f x0 = *(const v4f*)(in + i);
  v4f x1 = *(const v4f*)(in + i + 4);
  v8bf y;
  #pragma unroll
  for (int e = 0; e < 4; ++e) { y[e] = (__bf16)x0[e]; y[4 + e] = (__bf16)x1[e]; }
  *(v8bf*)(out + i) = y;
}

// ---------------------------------------------------------------------------
// Transpose + convert: in[R][C] f32 -> outT[C][R] bf16. 32x32 LDS tiles.
// ---------------------------------------------------------------------------
__global__ __launch_bounds__(256)
void transpose_f32_bf16(const float* __restrict__ in, __bf16* __restrict__ outT,
                        int R, int C) {
  __shared__ float tile[32][33];
  const int tr  = threadIdx.x >> 3;        // 0..31
  const int tc4 = (threadIdx.x & 7) * 4;   // 0..28
  const int rBase = blockIdx.y * 32;
  const int cBase = blockIdx.x * 32;

  v4f x = *(const v4f*)(in + (size_t)(rBase + tr) * C + (cBase + tc4));
  #pragma unroll
  for (int e = 0; e < 4; ++e) tile[tr][tc4 + e] = x[e];
  __syncthreads();

  v4bf y;
  #pragma unroll
  for (int e = 0; e < 4; ++e) y[e] = (__bf16)tile[tc4 + e][tr];
  *(v4bf*)(outT + (size_t)(cBase + tr) * R + (rBase + tc4)) = y;
}

// ---------------------------------------------------------------------------
// GEMM: C[M,N] = A[M,K] * Bt[N,K]^T, bf16 in, f32 out.
// Block tile 64x128, 256 threads = 8 waves; wave = (4 row tiles) x (2 col
// groups of 64). K-step 64: per stage each wave does 2 A-frags + 8 B-frags +
// 8 WMMA. Double-buffered LDS; staging is pure b128 copy. Fragments are
// preloaded into distinct registers so WMMA overlaps in-flight ds loads.
// ---------------------------------------------------------------------------
__global__ __launch_bounds__(256)
void gemm_bf16_nt(const __bf16* __restrict__ A, const __bf16* __restrict__ Bt,
                  float* __restrict__ C, int M, int N, int K) {
  __shared__ __align__(16) __bf16 As[2][64][72];    // 144B rows (16B aligned)
  __shared__ __align__(16) __bf16 Bs[2][128][72];

  const int tid   = threadIdx.x;
  const int lane  = tid & 31;
  const int wave  = tid >> 5;
  const int mBase = blockIdx.y * 64;
  const int nBase = blockIdx.x * 128;
  const int waveM = (wave & 3) * 16;   // 4 row tiles
  const int colG  = (wave >> 2) * 64;  // 2 col groups of 64

  v8f acc[4] = {};
  v8bf ra[2];   // staging registers: A = 2 x v8bf, B = 4 x v8bf per thread
  v8bf rb[4];

  const int am = waveM + (lane & 15);
  const int kb = (lane >> 4) * 8;
  const int bn = colG + (lane & 15);

  auto loadRegs = [&](int k0) {
    #pragma unroll
    for (int i = 0; i < 2; ++i) {
      int idx = tid + 256 * i;                 // 0..511
      int r = idx >> 3, c8 = (idx & 7) * 8;    // A: 64 rows x 64 k
      ra[i] = *(const v8bf*)(A + (size_t)(mBase + r) * K + (k0 + c8));
    }
    #pragma unroll
    for (int i = 0; i < 4; ++i) {
      int idx = tid + 256 * i;                 // 0..1023
      int r = idx >> 3, c8 = (idx & 7) * 8;    // B: 128 rows x 64 k
      rb[i] = *(const v8bf*)(Bt + (size_t)(nBase + r) * K + (k0 + c8));
    }
  };
  auto storeTile = [&](int buf) {
    #pragma unroll
    for (int i = 0; i < 2; ++i) {
      int idx = tid + 256 * i;
      int r = idx >> 3, c8 = (idx & 7) * 8;
      *(v8bf*)&As[buf][r][c8] = ra[i];
    }
    #pragma unroll
    for (int i = 0; i < 4; ++i) {
      int idx = tid + 256 * i;
      int r = idx >> 3, c8 = (idx & 7) * 8;
      *(v8bf*)&Bs[buf][r][c8] = rb[i];
    }
  };

  const int T = K / 64;
  loadRegs(0);
  storeTile(0);
  __syncthreads();

  for (int it = 0; it < T; ++it) {
    if (it + 1 < T) loadRegs((it + 1) * 64);   // issue next tile's loads

    const int cur = it & 1;
    const __bf16* arow = &As[cur][am][0];

    // Preload all fragments of the stage into distinct registers so the
    // scheduler can overlap v_wmma with in-flight ds_load_b128s.
    v16bf a0 = frag16(arow, kb);
    v16bf b00 = frag16(&Bs[cur][bn][0], kb);
    v16bf b01 = frag16(&Bs[cur][bn + 16][0], kb);
    v16bf b02 = frag16(&Bs[cur][bn + 32][0], kb);
    v16bf b03 = frag16(&Bs[cur][bn + 48][0], kb);
    v16bf a1 = frag16(arow + 32, kb);
    acc[0] = wmma_bf16(a0, b00, acc[0]);
    v16bf b10 = frag16(&Bs[cur][bn][32], kb);
    acc[1] = wmma_bf16(a0, b01, acc[1]);
    v16bf b11 = frag16(&Bs[cur][bn + 16][32], kb);
    acc[2] = wmma_bf16(a0, b02, acc[2]);
    v16bf b12 = frag16(&Bs[cur][bn + 32][32], kb);
    acc[3] = wmma_bf16(a0, b03, acc[3]);
    v16bf b13 = frag16(&Bs[cur][bn + 48][32], kb);
    acc[0] = wmma_bf16(a1, b10, acc[0]);
    acc[1] = wmma_bf16(a1, b11, acc[1]);
    acc[2] = wmma_bf16(a1, b12, acc[2]);
    acc[3] = wmma_bf16(a1, b13, acc[3]);

    if (it + 1 < T) storeTile((it + 1) & 1);   // s_wait_loadcnt sinks here
    __syncthreads();
  }

  // C/D layout: lane half selects row+8; n = lane&15
  const int rowOff = (lane >> 4) * 8;
  const int cn = lane & 15;
  #pragma unroll
  for (int j = 0; j < 4; ++j) {
    #pragma unroll
    for (int r = 0; r < 8; ++r) {
      int row = mBase + waveM + rowOff + r;
      C[(size_t)row * N + nBase + colG + j * 16 + cn] = acc[j][r];
    }
  }
}

// ---------------------------------------------------------------------------
// RoPE + split. Q/K seq-major (QK^T fragments d-contiguous).
// V written TRANSPOSED [kvh][d][seq] so PV fragments are kv-contiguous.
// ---------------------------------------------------------------------------
__global__ __launch_bounds__(64)
void rope_split(const int* __restrict__ pos, const float* __restrict__ qkv,
                __bf16* __restrict__ Qb, __bf16* __restrict__ Kb,
                __bf16* __restrict__ Vt) {
  const int s  = blockIdx.x;
  const int hh = blockIdx.y;
  const int d  = threadIdx.x;                 // 0..63
  const float p = (float)pos[s];
  const float inv_freq = __powf(10000.0f, -(float)d / 64.0f);
  float sn, cs;
  __sincosf(p * inv_freq, &sn, &cs);

  if (hh < NUM_HEADS) {
    const float* src = qkv + (size_t)s * QKV_OUT + hh * HEAD_DIM;
    __bf16* dst = Qb + ((size_t)hh * SEQ + s) * HEAD_DIM;
    float x1 = src[d], x2 = src[d + 64];
    dst[d]      = (__bf16)(x1 * cs - x2 * sn);
    dst[d + 64] = (__bf16)(x2 * cs + x1 * sn);
  } else if (hh < NUM_HEADS + NUM_KV_HEADS) {
    int h = hh - NUM_HEADS;
    const float* src = qkv + (size_t)s * QKV_OUT + Q_SIZE + h * HEAD_DIM;
    __bf16* dst = Kb + ((size_t)h * SEQ + s) * HEAD_DIM;
    float x1 = src[d], x2 = src[d + 64];
    dst[d]      = (__bf16)(x1 * cs - x2 * sn);
    dst[d + 64] = (__bf16)(x2 * cs + x1 * sn);
  } else {
    int h = hh - NUM_HEADS - NUM_KV_HEADS;
    const float* src = qkv + (size_t)s * QKV_OUT + Q_SIZE + KV_SIZE + h * HEAD_DIM;
    Vt[((size_t)h * HEAD_DIM + d)      * SEQ + s] = (__bf16)src[d];
    Vt[((size_t)h * HEAD_DIM + d + 64) * SEQ + s] = (__bf16)src[d + 64];
  }
}

// ---------------------------------------------------------------------------
// Causal flash attention, GQA. grid = (SEQ/64, NUM_HEADS), 128 threads.
// Block-uniform kv loop; K (32x128) and V^T (128x32) tiles staged in LDS once
// per tile, shared by all 4 waves. Output written directly in bf16 so the
// output-projection GEMM consumes it without conversion.
// ---------------------------------------------------------------------------
__global__ __launch_bounds__(128)
void attn_fwd(const __bf16* __restrict__ Qb, const __bf16* __restrict__ Kb,
              const __bf16* __restrict__ Vt, __bf16* __restrict__ O) {
  __shared__ __align__(16) __bf16 Ks[32][136];     // [kv][d], 272B rows
  __shared__ __align__(16) __bf16 Vs[128][40];     // [d][kv], 80B rows
  __shared__ __align__(16) __bf16 Plds[4][16][32]; // per-wave P staging

  const int tid   = threadIdx.x;
  const int lane  = tid & 31;
  const int wave  = tid >> 5;
  const int head  = blockIdx.y;
  const int kvh   = head / GROUP;
  const int qblk  = blockIdx.x * 64;
  const int qbase = qblk + wave * 16;
  const float scale = 0.08838834764831845f;  // 1/sqrt(128)

  const int am     = lane & 15;
  const int kb     = (lane >> 4) * 8;
  const int n15    = lane & 15;
  const int rowOff = (lane >> 4) * 8;

  // Preload Q fragments: 4 chunks of K=32 covering head_dim=128.
  v16bf aq[4];
  {
    const __bf16* qrow = Qb + ((size_t)head * SEQ + qbase + am) * HEAD_DIM;
    #pragma unroll
    for (int c = 0; c < 4; ++c) aq[c] = frag16(qrow + c * 32, kb);
  }

  float mrow[8], lrow[8];
  v8f o[8] = {};
  #pragma unroll
  for (int r = 0; r < 8; ++r) { mrow[r] = -INFINITY; lrow[r] = 0.0f; }

  const int kvEndBlk = qblk + 64;   // block-uniform causal extent
  for (int kv0 = 0; kv0 < kvEndBlk; kv0 += 32) {
    // ---- cooperative staging: K tile 32x128, V^T tile 128x32 ----
    #pragma unroll
    for (int i = 0; i < 4; ++i) {
      int idx = tid + 128 * i;            // 0..511
      int r = idx >> 4, c8 = (idx & 15) * 8;
      *(v8bf*)&Ks[r][c8] =
          *(const v8bf*)(Kb + ((size_t)kvh * SEQ + kv0 + r) * HEAD_DIM + c8);
    }
    #pragma unroll
    for (int i = 0; i < 4; ++i) {
      int idx = tid + 128 * i;            // 0..511
      int d = idx >> 2, c8 = (idx & 3) * 8;
      *(v8bf*)&Vs[d][c8] =
          *(const v8bf*)(Vt + ((size_t)kvh * HEAD_DIM + d) * SEQ + kv0 + c8);
    }
    __syncthreads();

    if (kv0 < qbase + 16) {   // wave-uniform: this wave has unmasked rows
      // ---- S(16x32) = Q(16x128) * K^T(128x32) ----
      v8f s0 = {};
      v8f s1 = {};
      #pragma unroll
      for (int c = 0; c < 4; ++c) {
        v16bf bk0 = frag16(&Ks[n15][c * 32], kb);
        v16bf bk1 = frag16(&Ks[n15 + 16][c * 32], kb);
        s0 = wmma_bf16(aq[c], bk0, s0);
        s1 = wmma_bf16(aq[c], bk1, s1);
      }

      // ---- online softmax (per row; shuffles stay in 16-lane half) ----
      #pragma unroll
      for (int r = 0; r < 8; ++r) {
        int g = qbase + rowOff + r;
        float v0 = ((kv0 + n15) <= g)      ? s0[r] * scale : -INFINITY;
        float v1 = ((kv0 + 16 + n15) <= g) ? s1[r] * scale : -INFINITY;
        float t = fmaxf(v0, v1);
        #pragma unroll
        for (int m = 8; m >= 1; m >>= 1) t = fmaxf(t, __shfl_xor(t, m, 32));
        float mnew  = fmaxf(mrow[r], t);
        float alpha = (mnew == -INFINITY) ? 1.0f : __expf(mrow[r] - mnew);
        float p0 = (v0 == -INFINITY) ? 0.0f : __expf(v0 - mnew);
        float p1 = (v1 == -INFINITY) ? 0.0f : __expf(v1 - mnew);
        float ps = p0 + p1;
        #pragma unroll
        for (int m = 8; m >= 1; m >>= 1) ps += __shfl_xor(ps, m, 32);
        lrow[r] = lrow[r] * alpha + ps;
        mrow[r] = mnew;
        #pragma unroll
        for (int j = 0; j < 8; ++j) o[j][r] *= alpha;
        Plds[wave][rowOff + r][n15]      = (__bf16)p0;
        Plds[wave][rowOff + r][n15 + 16] = (__bf16)p1;
      }
      asm volatile("s_wait_dscnt 0" ::: "memory");  // per-wave DS fence

      // ---- re-stripe P into A-fragment layout (two ds_load_b128) ----
      v16bf pa = frag16(&Plds[wave][am][0], kb);

      // ---- O(16x128) += P(16x32) * V(32x128) ----
      #pragma unroll
      for (int j = 0; j < 8; ++j) {
        v16bf bv = frag16(&Vs[j * 16 + n15][0], kb);
        o[j] = wmma_bf16(pa, bv, o[j]);
      }
    }
    __syncthreads();
  }

  // ---- epilogue: O / l, write bf16 [SEQ, NUM_HEADS*HEAD_DIM] ----
  #pragma unroll
  for (int r = 0; r < 8; ++r) {
    int g = qbase + rowOff + r;
    float inv = 1.0f / lrow[r];
    #pragma unroll
    for (int j = 0; j < 8; ++j)
      O[(size_t)g * Q_SIZE + head * HEAD_DIM + j * 16 + n15] =
          (__bf16)(o[j][r] * inv);
  }
}

// ---------------------------------------------------------------------------
// Launcher
// ---------------------------------------------------------------------------
extern "C" void kernel_launch(void* const* d_in, const int* in_sizes, int n_in,
                              void* d_out, int out_size, void* d_ws, size_t ws_size,
                              hipStream_t stream) {
  (void)in_sizes; (void)n_in; (void)out_size; (void)ws_size;

  const int*   positions = (const int*)d_in[0];
  const float* hidden    = (const float*)d_in[1];
  const float* Wqkv      = (const float*)d_in[2];
  const float* Wo        = (const float*)d_in[3];
  float*       out       = (float*)d_out;

  // Workspace layout (~136 MB) with deterministic intra-call aliasing:
  //   regionA: Hb (stage 1)  -> attnB (stages 3-4)
  //   regionB: WqkvT
  //   regionC: qkv f32 (stage 2) -> WoT (stage 4)
  //   regionD: Qb, Kb, Vt
  char* ws = (char*)d_ws;
  __bf16* Hb    = (__bf16*)ws;
  __bf16* attnB = (__bf16*)ws;  ws += (size_t)SEQ * HIDDEN * sizeof(__bf16);
  __bf16* WqkvT = (__bf16*)ws;  ws += (size_t)QKV_OUT * HIDDEN * sizeof(__bf16);
  float*  qkv   = (float*)ws;
  __bf16* WoT   = (__bf16*)ws;  ws += (size_t)SEQ * QKV_OUT * sizeof(float);
  __bf16* Qb    = (__bf16*)ws;  ws += (size_t)NUM_HEADS    * SEQ * HEAD_DIM * sizeof(__bf16);
  __bf16* Kb    = (__bf16*)ws;  ws += (size_t)NUM_KV_HEADS * SEQ * HEAD_DIM * sizeof(__bf16);
  __bf16* Vt    = (__bf16*)ws;

  // 0) precondition operands for the WMMA path
  convert_f32_bf16<<<(SEQ * HIDDEN) / (256 * 8), 256, 0, stream>>>(hidden, Hb);
  transpose_f32_bf16<<<dim3(QKV_OUT / 32, HIDDEN / 32), 256, 0, stream>>>(
      Wqkv, WqkvT, HIDDEN, QKV_OUT);

  // 1) qkv = hidden @ Wqkv   [2048,4096]x[4096,6144]
  gemm_bf16_nt<<<dim3(QKV_OUT / 128, SEQ / 64), 256, 0, stream>>>(
      Hb, WqkvT, qkv, SEQ, QKV_OUT, HIDDEN);

  // 2) RoPE + split (V transposed)
  rope_split<<<dim3(SEQ, NUM_HEADS + 2 * NUM_KV_HEADS), 64, 0, stream>>>(
      positions, qkv, Qb, Kb, Vt);

  // 3) causal flash attention (GQA x4) -> bf16
  attn_fwd<<<dim3(SEQ / 64, NUM_HEADS), 128, 0, stream>>>(Qb, Kb, Vt, attnB);

  // 4) transpose Wo into qkv's (now dead) region, then out = attn @ Wo
  transpose_f32_bf16<<<dim3(HIDDEN / 32, HIDDEN / 32), 256, 0, stream>>>(
      Wo, WoT, HIDDEN, HIDDEN);
  gemm_bf16_nt<<<dim3(HIDDEN / 128, SEQ / 64), 256, 0, stream>>>(
      attnB, WoT, out, SEQ, HIDDEN, HIDDEN);
}
